// MultiHeadAttentionComplex_2946347565893
// MI455X (gfx1250) — compile-verified, matrix operand order checked by reading, not run on hardware
//
#include <hip/hip_runtime.h>
#include <math.h>

// Problem constants (from reference): B=2, S=2048, D=1024, H=16, DK=64
#define BB  2
#define SS  2048
#define DD  1024
#define HH  16
#define DKK 64
static constexpr float ALPHA = 0.3f;

typedef __attribute__((ext_vector_type(16))) __bf16 bf16x16;
typedef __attribute__((ext_vector_type(8)))  __bf16 bf16x8;
typedef __attribute__((ext_vector_type(8)))  float  f32x8;
typedef __attribute__((ext_vector_type(4)))  float  f32x4;

// ---------------------------------------------------------------------------
// WMMA helper: D = A(16x32 bf16) * B(32x16 bf16) + C(16x16 f32)
// bf16 WMMA has no A/B negate (ISA NEG table: CNeg only) -> pass false.
// ---------------------------------------------------------------------------
__device__ __forceinline__ f32x8 wmma_bf16(bf16x16 a, bf16x16 b, f32x8 c) {
  return __builtin_amdgcn_wmma_f32_16x16x32_bf16(false, a, false, b,
                                                 (short)0, c, false, false);
}

// Load one lane's A-row / B-column fragment for a K=32 chunk.
// ISA 16-bit A layout: lanes 0-15 hold K=[0..7]+[16..23], lanes 16-31 hold
// K=[8..15]+[24..31] -> two contiguous 8-element (16B) runs per lane.
__device__ __forceinline__ bf16x16 load_frag_k32(const __bf16* p, int hh) {
  bf16x8 lo = *(const bf16x8*)(p + hh * 8);
  bf16x8 hi = *(const bf16x8*)(p + 16 + hh * 8);
  return __builtin_shufflevector(lo, hi, 0, 1, 2, 3, 4, 5, 6, 7,
                                 8, 9, 10, 11, 12, 13, 14, 15);
}

// ---------------------------------------------------------------------------
// fp32 -> bf16 elementwise convert (with sign, for the pre-negated -z_imag)
// ---------------------------------------------------------------------------
__global__ void k_f32_to_bf16(const float* __restrict__ in,
                              __bf16* __restrict__ out, float sgn, int n) {
  int i = blockIdx.x * blockDim.x + threadIdx.x;
  if (i < n) out[i] = (__bf16)(in[i] * sgn);
}

// Weight [K=D, N=D] fp32 row-major -> bf16 transposed [N, K] so B-operand
// columns become contiguous (two 16B loads per lane per K-chunk).
__global__ void k_w_to_bf16_t(const float* __restrict__ in,
                              __bf16* __restrict__ out) {
  int i = blockIdx.x * blockDim.x + threadIdx.x; // over D*D
  int n = i >> 10;        // output row (was column)
  int k = i & (DD - 1);   // contiguous in output
  out[(size_t)n * DD + k] = (__bf16)in[(size_t)k * DD + n];
}

// V [B,S,D] bf16 -> Vt [B,H,DK,S] bf16 (keys contiguous per dk for P@V B-op)
__global__ void k_transpose_v(const __bf16* __restrict__ v,
                              __bf16* __restrict__ vt) {
  int i = blockIdx.x * blockDim.x + threadIdx.x; // over B*S*D
  int d = i & (DD - 1);
  int s = (i >> 10) & (SS - 1);
  int b = i >> 21;
  int h = d >> 6, dk = d & (DKK - 1);
  vt[((size_t)(b * HH + h) * DKK + dk) * SS + s] = v[i];
}

// ---------------------------------------------------------------------------
// Tiled GEMM, 32x64 output tile per wave (2x4 grid of 16x16 WMMA tiles).
//   out = A1 @ W1t^T [+ A2 @ W2t^T]  (+ bias, fp32 out, for the final proj)
// A: [M,K] bf16 row-major; W*t: [N,K] bf16 (pre-transposed).
// DUAL/BIASED are compile-time -> no per-iteration branches in the K loop.
// Per K=32 chunk: 12 b128 loads feed 8 WMMAs (24/16 in DUAL mode).
// ---------------------------------------------------------------------------
template <bool DUAL, bool BIASED, typename OutT>
__global__ void k_gemm_tile(const __bf16* __restrict__ A1,
                            const __bf16* __restrict__ W1,
                            const __bf16* __restrict__ A2,
                            const __bf16* __restrict__ W2,
                            const float* __restrict__ bias,
                            OutT* __restrict__ out, int M, int N, int K) {
  const int wid = (blockIdx.x * blockDim.x + threadIdx.x) >> 5;
  const int lane = threadIdx.x & 31;
  const int tilesN = N >> 6; // 64-wide N tiles
  const int tm = wid / tilesN, tn = wid - tm * tilesN;
  const int m16 = lane & 15, hh = lane >> 4;

  const __bf16* aA[2];
  const __bf16* aW[4];
#pragma unroll
  for (int i = 0; i < 2; ++i)
    aA[i] = A1 + (size_t)(tm * 32 + i * 16 + m16) * K;
#pragma unroll
  for (int j = 0; j < 4; ++j)
    aW[j] = W1 + (size_t)(tn * 64 + j * 16 + m16) * K;

  const __bf16* bA[2];
  const __bf16* bW[4];
  if constexpr (DUAL) {
#pragma unroll
    for (int i = 0; i < 2; ++i)
      bA[i] = A2 + (size_t)(tm * 32 + i * 16 + m16) * K;
#pragma unroll
    for (int j = 0; j < 4; ++j)
      bW[j] = W2 + (size_t)(tn * 64 + j * 16 + m16) * K;
  }

  f32x8 acc[2][4];
#pragma unroll
  for (int i = 0; i < 2; ++i)
#pragma unroll
    for (int j = 0; j < 4; ++j)
      acc[i][j] = f32x8{0.f, 0.f, 0.f, 0.f, 0.f, 0.f, 0.f, 0.f};

  for (int kc = 0; kc < K; kc += 32) {
    // stream-ahead hint for the activation rows (global_prefetch_b8)
    __builtin_prefetch(aA[0] + kc + 512, 0, 1);
    __builtin_prefetch(aA[1] + kc + 512, 0, 1);
    bf16x16 fa[2], fw[4];
#pragma unroll
    for (int i = 0; i < 2; ++i) fa[i] = load_frag_k32(aA[i] + kc, hh);
#pragma unroll
    for (int j = 0; j < 4; ++j) fw[j] = load_frag_k32(aW[j] + kc, hh);
#pragma unroll
    for (int i = 0; i < 2; ++i)
#pragma unroll
      for (int j = 0; j < 4; ++j)
        acc[i][j] = wmma_bf16(fa[i], fw[j], acc[i][j]);
    if constexpr (DUAL) {
      bf16x16 ga[2], gw[4];
#pragma unroll
      for (int i = 0; i < 2; ++i) ga[i] = load_frag_k32(bA[i] + kc, hh);
#pragma unroll
      for (int j = 0; j < 4; ++j) gw[j] = load_frag_k32(bW[j] + kc, hh);
#pragma unroll
      for (int i = 0; i < 2; ++i)
#pragma unroll
        for (int j = 0; j < 4; ++j)
          acc[i][j] = wmma_bf16(ga[i], gw[j], acc[i][j]);
    }
  }

  // C/D layout: lane holds col n=m16, rows r+8*hh
#pragma unroll
  for (int i = 0; i < 2; ++i) {
#pragma unroll
    for (int j = 0; j < 4; ++j) {
      const size_t ob =
          (size_t)(tm * 32 + i * 16) * N + (size_t)tn * 64 + j * 16 + m16;
#pragma unroll
      for (int r = 0; r < 8; ++r) {
        if constexpr (BIASED) {
          out[ob + (size_t)(r + 8 * hh) * N] =
              acc[i][j][r] + bias[tn * 64 + j * 16 + m16];
        } else {
          out[ob + (size_t)(r + 8 * hh) * N] = (OutT)acc[i][j][r];
        }
      }
    }
  }
}

// ---------------------------------------------------------------------------
// Attention: one 256-thread (8 wave32) workgroup per (b, 16-query block).
// Loops heads; LDS holds the 16 x 2048 fp32 score strip (128KB, dynamic LDS
// -- CDNA5 allows up to 320KB/workgroup).
//   Phase A: Sr/Si score tiles via bf16 WMMA with hoisted Q fragments;
//            hybrid score uses one v_rsq (mag = s2*rsq, cos = sr*rsq).
//   Phase B: float4-vectorized row softmax + probs head-mean (b128 traffic).
//   Phase C: context = P @ V, K split across all 8 waves + LDS combine.
// ---------------------------------------------------------------------------
__global__ void k_attn(const __bf16* __restrict__ Qr, const __bf16* __restrict__ Qi,
                       const __bf16* __restrict__ Kr, const __bf16* __restrict__ Ki,
                       const __bf16* __restrict__ Vt, const int* __restrict__ mask,
                       __bf16* __restrict__ ctx, float* __restrict__ pmean) {
  extern __shared__ float sm[];
  float* sc = sm;                  // 16 * 2048 score strip
  float* red = sm + 16 * SS;       // 16 * 16 reduction scratch
  float* rowstat = red + 256;      // [0..15]=rowmax, [16..31]=1/rowsum
  float* part = rowstat + 32;      // 4 * 256 phase-C partial tiles

  const int tid = threadIdx.x;
  const int lane = tid & 31, wave = tid >> 5;
  const int m16 = lane & 15, hh = lane >> 4;
  const int b = blockIdx.x / (SS / 16);
  const int q0 = (blockIdx.x % (SS / 16)) * 16;
  const float scale = 0.125f; // 1/sqrt(DK)

  for (int h = 0; h < HH; ++h) {
    // ---------------- Phase A: hybrid scores into LDS ----------------
    const __bf16* qr_row = Qr + ((size_t)(b * SS + q0 + m16) * DD + h * DKK);
    const __bf16* qi_row = Qi + ((size_t)(b * SS + q0 + m16) * DD + h * DKK);
    // Q fragments are invariant over the key loop: load once per head.
    const bf16x16 fqr0 = load_frag_k32(qr_row, hh);
    const bf16x16 fqr1 = load_frag_k32(qr_row + 32, hh);
    const bf16x16 fqi0 = load_frag_k32(qi_row, hh);
    const bf16x16 fqi1 = load_frag_k32(qi_row + 32, hh);

#pragma unroll 2
    for (int kt = wave; kt < SS / 16; kt += 8) {
      const int k0 = kt * 16;
      const __bf16* kr_col = Kr + ((size_t)(b * SS + k0 + m16) * DD + h * DKK);
      const __bf16* ki_col = Ki + ((size_t)(b * SS + k0 + m16) * DD + h * DKK);
      // prefetch this wave's next key tile (128 rows ahead)
      __builtin_prefetch(kr_col + (size_t)128 * DD, 0, 1);
      __builtin_prefetch(ki_col + (size_t)128 * DD, 0, 1);

      const bf16x16 bkr0 = load_frag_k32(kr_col, hh);
      const bf16x16 bkr1 = load_frag_k32(kr_col + 32, hh);
      const bf16x16 bki0 = load_frag_k32(ki_col, hh);
      const bf16x16 bki1 = load_frag_k32(ki_col + 32, hh);

      f32x8 accr = {0.f, 0.f, 0.f, 0.f, 0.f, 0.f, 0.f, 0.f};
      f32x8 ai1  = {0.f, 0.f, 0.f, 0.f, 0.f, 0.f, 0.f, 0.f};
      f32x8 ai2  = {0.f, 0.f, 0.f, 0.f, 0.f, 0.f, 0.f, 0.f};
      accr = wmma_bf16(fqr0, bkr0, accr);  // Sr += Qr.Kr
      accr = wmma_bf16(fqi0, bki0, accr);  // Sr += Qi.Ki
      ai1  = wmma_bf16(fqi0, bkr0, ai1);   // Si += Qi.Kr
      ai2  = wmma_bf16(fqr0, bki0, ai2);   // Si -= Qr.Ki (VALU sub below)
      accr = wmma_bf16(fqr1, bkr1, accr);
      accr = wmma_bf16(fqi1, bki1, accr);
      ai1  = wmma_bf16(fqi1, bkr1, ai1);
      ai2  = wmma_bf16(fqr1, bki1, ai2);

      const int kk = k0 + m16;
      const bool valid = mask[b * SS + kk] != 0;
#pragma unroll
      for (int r = 0; r < 8; ++r) {
        float sr = accr[r] * scale;
        float si = (ai1[r] - ai2[r]) * scale;
        // mag = sqrt(s2) = s2*rsq;  cos = sr/mag = sr*rsq  (one v_rsq_f32)
        float s2 = sr * sr + si * si;
        float rsq = __builtin_amdgcn_rsqf(s2);
        float v = (s2 * rsq + ALPHA * (sr * rsq)) * scale;
        sc[(r + 8 * hh) * SS + kk] = valid ? v : -1e9f;
      }
    }
    __syncthreads();

    // ------- Phase B: float4 softmax + probs head-mean (b128 traffic) -------
    {
      const int row = tid >> 4, j = tid & 15;
      f32x4* srow4 = (f32x4*)(sc + row * SS);           // 512 float4
      f32x4 mv = {-3.4e38f, -3.4e38f, -3.4e38f, -3.4e38f};
      for (int k4 = j; k4 < SS / 4; k4 += 16) {
        f32x4 v = srow4[k4];
#pragma unroll
        for (int e = 0; e < 4; ++e) mv[e] = fmaxf(mv[e], v[e]);
      }
      red[row * 16 + j] =
          fmaxf(fmaxf(mv[0], mv[1]), fmaxf(mv[2], mv[3]));
      __syncthreads();
      if (j == 0) {
        float m2 = red[row * 16];
#pragma unroll
        for (int t = 1; t < 16; ++t) m2 = fmaxf(m2, red[row * 16 + t]);
        rowstat[row] = m2;
      }
      __syncthreads();
      const float rmax = rowstat[row];
      float sum = 0.f;
      for (int k4 = j; k4 < SS / 4; k4 += 16) {
        f32x4 v = srow4[k4];
        f32x4 e;
#pragma unroll
        for (int t = 0; t < 4; ++t) e[t] = __expf(v[t] - rmax);
        srow4[k4] = e;
        sum += (e[0] + e[1]) + (e[2] + e[3]);
      }
      red[row * 16 + j] = sum;
      __syncthreads();
      if (j == 0) {
        float s2 = 0.f;
#pragma unroll
        for (int t = 0; t < 16; ++t) s2 += red[row * 16 + t];
        rowstat[16 + row] = 1.f / s2;
      }
      __syncthreads();
      const float inv = rowstat[16 + row];
      f32x4* pm4 = (f32x4*)(pmean + ((size_t)b * SS + (q0 + row)) * SS);
      for (int k4 = j; k4 < SS / 4; k4 += 16) {
        f32x4 p = srow4[k4] * inv;
        srow4[k4] = p;                         // normalized probs for P@V
        f32x4 add = p * (1.f / HH);
        if (h == 0) pm4[k4] = add;             // block owns this slice
        else        pm4[k4] = pm4[k4] + add;
      }
    }
    __syncthreads();

    // ---------- Phase C: context = P @ V (K split over 8 waves) ----------
    {
      const int ntile = wave & 3;            // 16-wide dk tile in this head
      const int khalf = wave >> 2;           // key range half
      const __bf16* vrow =
          Vt + ((size_t)(b * HH + h) * DKK + ntile * 16 + m16) * SS;
      f32x8 acc = {0.f, 0.f, 0.f, 0.f, 0.f, 0.f, 0.f, 0.f};
      const int kbeg = khalf * (SS / 2), kend = kbeg + SS / 2;
      for (int kc = kbeg; kc < kend; kc += 32) {
        // probs fragment from LDS: two 8-float contiguous runs -> 4x b128
        const int b0 = m16 * SS + kc + hh * 8;
        const f32x4 u0 = *(const f32x4*)(sc + b0);
        const f32x4 u1 = *(const f32x4*)(sc + b0 + 4);
        const f32x4 u2 = *(const f32x4*)(sc + b0 + 16);
        const f32x4 u3 = *(const f32x4*)(sc + b0 + 20);
        bf16x16 ap;
#pragma unroll
        for (int e = 0; e < 4; ++e) {
          ap[e]      = (__bf16)u0[e];
          ap[e + 4]  = (__bf16)u1[e];
          ap[e + 8]  = (__bf16)u2[e];
          ap[e + 12] = (__bf16)u3[e];
        }
        bf16x16 bv = load_frag_k32(vrow + kc, hh);
        acc = wmma_bf16(ap, bv, acc);
      }
      // upper-half waves stage their partial tiles in LDS
      if (wave >= 4) {
#pragma unroll
        for (int r = 0; r < 8; ++r)
          part[ntile * 256 + (r + 8 * hh) * 16 + m16] = acc[r];
      }
      __syncthreads();
      if (wave < 4) {
        const size_t cb =
            (size_t)(b * SS + q0) * DD + h * DKK + ntile * 16 + m16;
#pragma unroll
        for (int r = 0; r < 8; ++r) {
          float v = acc[r] + part[ntile * 256 + (r + 8 * hh) * 16 + m16];
          ctx[cb + (size_t)(r + 8 * hh) * DD] = (__bf16)v;
        }
      }
    }
    __syncthreads();   // protect sc/part before next head overwrites
  }
}

// ---------------------------------------------------------------------------
extern "C" void kernel_launch(void* const* d_in, const int* in_sizes, int n_in,
                              void* d_out, int out_size, void* d_ws, size_t ws_size,
                              hipStream_t stream) {
  (void)in_sizes; (void)n_in; (void)out_size; (void)ws_size;

  const float* z_real = (const float*)d_in[0];
  const float* z_imag = (const float*)d_in[1];
  const float* Wq_r   = (const float*)d_in[2];
  const float* Wq_i   = (const float*)d_in[3];
  const float* Wk_r   = (const float*)d_in[4];
  const float* Wk_i   = (const float*)d_in[5];
  const float* Wv     = (const float*)d_in[6];
  const float* Wo_w   = (const float*)d_in[7];
  const float* Wo_b   = (const float*)d_in[8];
  const int*   mask   = (const int*)d_in[9];

  float* out   = (float*)d_out;                          // [B,S,D]
  float* pmean = out + (size_t)BB * SS * DD;             // [B,S,S]

  // Workspace carving (bf16 staging buffers), 256B aligned
  size_t off = 0;
  auto carve = [&](size_t bytes) -> char* {
    char* p = (char*)d_ws + off;
    off += (bytes + 255) & ~(size_t)255;
    return p;
  };
  const size_t actN = (size_t)BB * SS * DD;   // 4,194,304
  const size_t wN   = (size_t)DD * DD;        // 1,048,576

  __bf16* zr  = (__bf16*)carve(actN * 2);
  __bf16* zi  = (__bf16*)carve(actN * 2);
  __bf16* zin = (__bf16*)carve(actN * 2);     // -z_imag (bf16 WMMA lacks neg)
  __bf16* wqr = (__bf16*)carve(wN * 2);
  __bf16* wqi = (__bf16*)carve(wN * 2);
  __bf16* wkr = (__bf16*)carve(wN * 2);
  __bf16* wki = (__bf16*)carve(wN * 2);
  __bf16* wv  = (__bf16*)carve(wN * 2);
  __bf16* wo  = (__bf16*)carve(wN * 2);
  __bf16* Qr  = (__bf16*)carve(actN * 2);
  __bf16* Qi  = (__bf16*)carve(actN * 2);
  __bf16* Krm = (__bf16*)carve(actN * 2);
  __bf16* Kim = (__bf16*)carve(actN * 2);
  __bf16* Vb  = (__bf16*)carve(actN * 2);
  __bf16* Vt  = (__bf16*)carve(actN * 2);
  __bf16* Ctx = (__bf16*)carve(actN * 2);

  const dim3 tpb(256);

  // 1) converts
  k_f32_to_bf16<<<(int)(actN / 256), tpb, 0, stream>>>(z_real, zr,  1.f, (int)actN);
  k_f32_to_bf16<<<(int)(actN / 256), tpb, 0, stream>>>(z_imag, zi,  1.f, (int)actN);
  k_f32_to_bf16<<<(int)(actN / 256), tpb, 0, stream>>>(z_imag, zin, -1.f, (int)actN);
  k_w_to_bf16_t<<<(int)(wN / 256), tpb, 0, stream>>>(Wq_r, wqr);
  k_w_to_bf16_t<<<(int)(wN / 256), tpb, 0, stream>>>(Wq_i, wqi);
  k_w_to_bf16_t<<<(int)(wN / 256), tpb, 0, stream>>>(Wk_r, wkr);
  k_w_to_bf16_t<<<(int)(wN / 256), tpb, 0, stream>>>(Wk_i, wki);
  k_w_to_bf16_t<<<(int)(wN / 256), tpb, 0, stream>>>(Wv,   wv);
  k_w_to_bf16_t<<<(int)(wN / 256), tpb, 0, stream>>>(Wo_w, wo);

  // 2) complex projections (M=4096, N=K=1024), 32x64 tile per wave:
  //    (M/32)*(N/64) = 2048 waves -> 256 blocks of 8 waves
  const int M = BB * SS, N = DD, K = DD;
  const int gblocks = (M / 32) * (N / 64) / 8; // 256
  k_gemm_tile<true, false, __bf16><<<gblocks, tpb, 0, stream>>>(
      zr, wqr, zin, wqi, nullptr, Qr, M, N, K);
  k_gemm_tile<true, false, __bf16><<<gblocks, tpb, 0, stream>>>(
      zi, wqr, zr, wqi, nullptr, Qi, M, N, K);
  k_gemm_tile<true, false, __bf16><<<gblocks, tpb, 0, stream>>>(
      zr, wkr, zin, wki, nullptr, Krm, M, N, K);
  k_gemm_tile<true, false, __bf16><<<gblocks, tpb, 0, stream>>>(
      zi, wkr, zr, wki, nullptr, Kim, M, N, K);
  k_gemm_tile<false, false, __bf16><<<gblocks, tpb, 0, stream>>>(
      zr, wv, nullptr, nullptr, nullptr, Vb, M, N, K);

  // 3) V -> [B,H,DK,S]
  k_transpose_v<<<(int)(actN / 256), tpb, 0, stream>>>(Vb, Vt);

  // 4) attention (dynamic LDS: score strip + reductions + phase-C staging)
  const size_t attn_lds = (16 * SS + 256 + 32 + 4 * 256) * sizeof(float);
  k_attn<<<BB * (SS / 16), tpb, attn_lds, stream>>>(Qr, Qi, Krm, Kim, Vt, mask,
                                                    Ctx, pmean);

  // 5) output projection + bias -> fp32 out
  k_gemm_tile<false, true, float><<<gblocks, tpb, 0, stream>>>(
      Ctx, wo, nullptr, nullptr, Wo_b, out, M, N, K);
}
